// ArcDecoder_6708738916355
// MI455X (gfx1250) — matrix operand order repaired and checked by reading, compiled, not run on hardware
//
#include <hip/hip_runtime.h>

typedef __attribute__((ext_vector_type(16))) _Float16 v16h;
typedef __attribute__((ext_vector_type(8)))  _Float16 v8h;
typedef __attribute__((ext_vector_type(8)))  float    v8f;

struct alignas(8) h4 { _Float16 v[4]; };

// ---------------------------------------------------------------------------
// Reduce within 16-lane half-waves (xor masks < 16 stay inside each half).
static __device__ __forceinline__ float red16(float x) {
    x += __shfl_xor(x, 1);
    x += __shfl_xor(x, 2);
    x += __shfl_xor(x, 4);
    x += __shfl_xor(x, 8);
    return x;
}
// Reduce across the full wave32.
static __device__ __forceinline__ float red32(float x) {
    x = red16(x);
    x += __shfl_xor(x, 16);
    return x;
}

// ---------------------------------------------------------------------------
// Convert weights: W1,W2 -> f16 (row-major, row = output neuron, so B[k][n] =
// W[n][k] is contiguous in k per lane), bilW -> f16 TRANSPOSED (bwt[n][k] =
// bilW[k][n]) so the bilinear GEMM uses the same contiguous B-fragment load.
__global__ __launch_bounds__(256) void convert_weights(
    const float* __restrict__ w1, const float* __restrict__ w2,
    const float* __restrict__ bw,
    _Float16* __restrict__ w1h, _Float16* __restrict__ w2h,
    _Float16* __restrict__ bwt) {
    int idx = blockIdx.x * 256 + threadIdx.x;
    if (idx < 16384) {
        w1h[idx] = (_Float16)w1[idx];
    } else if (idx < 32768) {
        int i = idx - 16384;
        w2h[i] = (_Float16)w2[i];
    } else if (idx < 49152) {
        int i = idx - 32768;           // i = n*128 + k
        int n = i >> 7, k = i & 127;
        bwt[i] = (_Float16)bw[k * 128 + n];
    }
}

// ---------------------------------------------------------------------------
// LayerNorm of z rows -> f16. One wave32 per row (4 floats per lane).
__global__ __launch_bounds__(256) void ln_rows(
    const float* __restrict__ z, const float* __restrict__ nw,
    const float* __restrict__ nb, _Float16* __restrict__ znh, int N) {
    const int wave = threadIdx.x >> 5, lane = threadIdx.x & 31;
    const int row = blockIdx.x * 8 + wave;
    if (row >= N) return;
    const float4 x = *reinterpret_cast<const float4*>(z + (size_t)row * 128 + lane * 4);
    float s1 = x.x + x.y + x.z + x.w;
    float s2 = x.x * x.x + x.y * x.y + x.z * x.z + x.w * x.w;
    s1 = red32(s1);
    s2 = red32(s2);
    const float mean = s1 * (1.0f / 128.0f);
    const float rs = rsqrtf(s2 * (1.0f / 128.0f) - mean * mean + 1e-5f);
    const int c = lane * 4;
    h4 o;
    o.v[0] = (_Float16)((x.x - mean) * rs * nw[c + 0] + nb[c + 0]);
    o.v[1] = (_Float16)((x.y - mean) * rs * nw[c + 1] + nb[c + 1]);
    o.v[2] = (_Float16)((x.z - mean) * rs * nw[c + 2] + nb[c + 2]);
    o.v[3] = (_Float16)((x.w - mean) * rs * nw[c + 3] + nb[c + 3]);
    *reinterpret_cast<h4*>(znh + (size_t)row * 128 + c) = o;
}

// ---------------------------------------------------------------------------
// GEMM: OUT[M x 128] = A[M x 128] (f16) x B[128 x 128], B[k][n] = W[n*128+k].
// One wave per 16-row tile; 8 column tiles x 4 K-steps of wmma_f32_16x16x32_f16.
// MODE 0: epilogue bias+ReLU+LayerNorm, store f16.
// MODE 1: epilogue bias+ReLU+LayerNorm, store f32.
// MODE 2: plain store f32 (bilinear projection).
template <int MODE>
__global__ __launch_bounds__(256) void gemm128(
    const _Float16* __restrict__ A, const _Float16* __restrict__ W,
    const float* __restrict__ bias, const float* __restrict__ nw,
    const float* __restrict__ nb, _Float16* __restrict__ outH,
    float* __restrict__ outF, int nTiles) {
    const int wave = threadIdx.x >> 5, lane = threadIdx.x & 31;
    const int tile = blockIdx.x * 8 + wave;
    if (tile >= nTiles) return;           // wave-uniform: EXEC stays all-ones
    const int l15 = lane & 15;
    const int hi = (lane >> 4) & 1;

    // A fragments: 16-bit A layout. Lane holds row m = tile*16 + (lane&15);
    // halves 0..15 = K kstep*32 + (hi?16:0)+[0..15] (contiguous).
    const _Float16* arow = A + (size_t)(tile * 16 + l15) * 128 + hi * 16;
    v16h afrag[4];
#pragma unroll
    for (int k = 0; k < 4; ++k) {
        v8h lo = *reinterpret_cast<const v8h*>(arow + k * 32);
        v8h hh = *reinterpret_cast<const v8h*>(arow + k * 32 + 8);
        afrag[k] = __builtin_shufflevector(lo, hh, 0, 1, 2, 3, 4, 5, 6, 7,
                                           8, 9, 10, 11, 12, 13, 14, 15);
    }

    v8f acc[8];
#pragma unroll
    for (int j = 0; j < 8; ++j) {
        v8f c = {0.f, 0.f, 0.f, 0.f, 0.f, 0.f, 0.f, 0.f};
        // B fragment: lane holds n = j*16 + (lane&15); halves = 16 contiguous
        // K values starting at kstep*32 + (hi?16:0)  ->  W row-major read.
        const _Float16* wrow = W + (size_t)(j * 16 + l15) * 128 + hi * 16;
#pragma unroll
        for (int k = 0; k < 4; ++k) {
            v16h b = *reinterpret_cast<const v16h*>(wrow + k * 32);
            c = __builtin_amdgcn_wmma_f32_16x16x32_f16(
                false, afrag[k], false, b, (short)0, c, false, false);
        }
        acc[j] = c;
    }

    // C/D layout: acc[j][r] is element (m = tile*16 + r + hi*8, n = j*16+l15).
#pragma unroll
    for (int r = 0; r < 8; ++r) {
        const size_t m = (size_t)(tile * 16 + r + hi * 8);
        if (MODE == 2) {
#pragma unroll
            for (int j = 0; j < 8; ++j)
                outF[m * 128 + j * 16 + l15] = acc[j][r];
        } else {
            float v[8], s1 = 0.f, s2 = 0.f;
#pragma unroll
            for (int j = 0; j < 8; ++j) {
                const int n = j * 16 + l15;
                float x = acc[j][r] + bias[n];
                x = fmaxf(x, 0.f);
                v[j] = x;
                s1 += x;
                s2 += x * x;
            }
            s1 = red16(s1);   // row m lives in one 16-lane half-wave
            s2 = red16(s2);
            const float mean = s1 * (1.0f / 128.0f);
            const float rs =
                rsqrtf(s2 * (1.0f / 128.0f) - mean * mean + 1e-5f);
#pragma unroll
            for (int j = 0; j < 8; ++j) {
                const int n = j * 16 + l15;
                const float y = (v[j] - mean) * rs * nw[n] + nb[n];
                if (MODE == 0)
                    outH[m * 128 + n] = (_Float16)y;
                else
                    outF[m * 128 + n] = y;
            }
        }
    }
}

// ---------------------------------------------------------------------------
// score[e] = dot(C[arc0], B[arc1]) + bilb. One wave32 per edge; L2-resident.
__global__ __launch_bounds__(256) void edge_score(
    const float* __restrict__ C, const float* __restrict__ B,
    const int* __restrict__ arcs, const float* __restrict__ bilb,
    float* __restrict__ out, int E) {
    const int lane = threadIdx.x & 31;
    const int e = (int)((blockIdx.x * 256 + threadIdx.x) >> 5);
    if (e >= E) return;
    const int a0 = arcs[2 * e];
    const int a1 = arcs[2 * e + 1];
    const float4 c4 =
        *reinterpret_cast<const float4*>(C + (size_t)a0 * 128 + lane * 4);
    const float4 b4 =
        *reinterpret_cast<const float4*>(B + (size_t)a1 * 128 + lane * 4);
    float s = c4.x * b4.x + c4.y * b4.y + c4.z * b4.z + c4.w * b4.w;
    s = red32(s);
    if (lane == 0) out[e] = s + bilb[0];
}

// ---------------------------------------------------------------------------
extern "C" void kernel_launch(void* const* d_in, const int* in_sizes, int n_in,
                              void* d_out, int out_size, void* d_ws,
                              size_t ws_size, hipStream_t stream) {
    const float* z   = (const float*)d_in[0];
    const int*   arc = (const int*)d_in[1];
    const float* l1w = (const float*)d_in[2];
    const float* l1b = (const float*)d_in[3];
    const float* l2w = (const float*)d_in[4];
    const float* l2b = (const float*)d_in[5];
    const float* bw  = (const float*)d_in[6];
    const float* bb  = (const float*)d_in[7];
    const float* nw  = (const float*)d_in[8];
    const float* nb  = (const float*)d_in[9];

    const int N = in_sizes[0] / 128;
    const int E = in_sizes[1] / 2;
    const int nTiles = (N + 15) / 16;
    const int Np = nTiles * 16;      // padded rows so WMMA tails stay in-bounds

    char* ws = (char*)d_ws;
    auto al = [](size_t x) { return (x + 255) & ~(size_t)255; };
    size_t off = 0;
    _Float16* znh = (_Float16*)(ws + off); off = al(off + (size_t)Np * 128 * 2);
    _Float16* h1h = (_Float16*)(ws + off); off = al(off + (size_t)Np * 128 * 2);
    float*    Bm  = (float*)(ws + off);    off = al(off + (size_t)Np * 128 * 4);
    float*    Cm  = (float*)(ws + off);    off = al(off + (size_t)Np * 128 * 4);
    _Float16* w1h = (_Float16*)(ws + off); off = al(off + 128 * 128 * 2);
    _Float16* w2h = (_Float16*)(ws + off); off = al(off + 128 * 128 * 2);
    _Float16* bwt = (_Float16*)(ws + off); off = al(off + 128 * 128 * 2);
    (void)ws_size; (void)n_in; (void)out_size;

    convert_weights<<<192, 256, 0, stream>>>(l1w, l2w, bw, w1h, w2h, bwt);
    ln_rows<<<(N + 7) / 8, 256, 0, stream>>>(z, nw, nb, znh, N);

    const int gb = (nTiles + 7) / 8;
    // h1 = LN(relu(zn@W1^T+b1)) -> f16
    gemm128<0><<<gb, 256, 0, stream>>>(znh, w1h, l1b, nw, nb, h1h, nullptr, nTiles);
    // h2 = LN(relu(zn@W2^T+b2)) -> f32
    gemm128<1><<<gb, 256, 0, stream>>>(znh, w2h, l2b, nw, nb, nullptr, Bm, nTiles);
    // C = h1 @ bilW -> f32 (B loaded from transposed bilW)
    gemm128<2><<<gb, 256, 0, stream>>>(h1h, bwt, nullptr, nullptr, nullptr, nullptr, Cm, nTiles);

    edge_score<<<(E + 7) / 8, 256, 0, stream>>>(Cm, Bm, arc, bb, (float*)d_out, E);
}
